// Block_28527172780354
// MI455X (gfx1250) — matrix operand order repaired and checked by reading, not compiled, not run on hardware
//
#include <hip/hip_runtime.h>
#include <hip/hip_bf16.h>

// Transformer block forward for MI455X (gfx1250, wave32, WMMA).
// One workgroup (8 waves) per batch element. All matmuls via
// v_wmma_f32_16x16x32_f16 (f16 in, f32 accumulate).
// Weights pre-packed once per launch into B-operand tile layout (f16) in the
// workspace -> hot loops are global_load_b128 + ds_load_b128 + v_wmma only.
// Register budget kept < 256 VGPRs (no s_set_vgpr_msb, no spills): A tiles
// are re-loaded from LDS per k-step instead of hoisted.

#define TT   64
#define CC   256
#define NH   16
#define HD   32
#define FFD  1024
#define SCALE_QK 0.17677669529663687f
#define LN_EPS   1e-5f

// Workspace layout (in _Float16 elements)
#define WQP_OFF 0
#define WKP_OFF 131072     // 16 heads * 8kk * 32n * 32ki
#define WVP_OFF 262144
#define WPP_OFF 393216     // 16kk * 256n * 32ki
#define W1P_OFF 524288     // 8kk * 1024n * 32ki
#define W2P_OFF 786432     // 32kk * 256n * 32ki

typedef __attribute__((ext_vector_type(16))) _Float16 v16h;
typedef __attribute__((ext_vector_type(8)))  _Float16 v8h;
typedef __attribute__((ext_vector_type(2)))  _Float16 v2h;
typedef __attribute__((ext_vector_type(8)))  float    v8f;

__device__ __forceinline__ v8f wmma16(v16h a, v16h b, v8f c) {
  return __builtin_amdgcn_wmma_f32_16x16x32_f16(false, a, false, b,
                                                (short)0, c, false, false);
}

// ---- weight pre-pack: [K x N] f32 (row-major) -> pack[kk][n][ki] f16 ----
__global__ void pack_weights_kernel(const float* __restrict__ src,
                                    _Float16* __restrict__ dst, int K, int N,
                                    size_t srcStride, size_t dstStride) {
  const float* s = src + (size_t)blockIdx.y * srcStride;
  _Float16* d = dst + (size_t)blockIdx.y * dstStride;
  const int total = K * N;
  for (int idx = blockIdx.x * blockDim.x + threadIdx.x; idx < total;
       idx += gridDim.x * blockDim.x) {
    int k = idx / N, n = idx - k * N;
    d[((size_t)(k >> 5) * N + n) * 32 + (k & 31)] = (_Float16)s[idx];
  }
}

// A operand (16x32 f16 tile) from row-major [M x ld] f16 LDS buffer.
// Lane m=lane%16, half h=lane/16; elems 0-15 -> K=16h+e.
__device__ __forceinline__ v16h load_A_lds(const _Float16* buf, int m0, int k0,
                                           int ld, int lane) {
  int m = m0 + (lane & 15);
  int h = lane >> 4;
  const _Float16* p = buf + m * ld + k0 + 16 * h;
  v8h lo = *(const v8h*)(p);
  v8h hi = *(const v8h*)(p + 8);
  v16h r;
#pragma unroll
  for (int e = 0; e < 8; ++e) { r[e] = lo[e]; r[e + 8] = hi[e]; }
  return r;
}

// B operand where B[k][n] = buf[n][k], buf row-major [N x ld] f16 LDS.
__device__ __forceinline__ v16h load_Bt_lds(const _Float16* buf, int n0, int k0,
                                            int ld, int lane) {
  int n = n0 + (lane & 15);
  int h = lane >> 4;
  const _Float16* p = buf + n * ld + k0 + 16 * h;
  v8h a = *(const v8h*)(p);
  v8h b = *(const v8h*)(p + 8);
  v16h r;
#pragma unroll
  for (int e = 0; e < 8; ++e) { r[e] = a[e]; r[e + 8] = b[e]; }
  return r;
}

// B operand from pre-packed weights: pack[kk][n][ki]; lane reads 32B contig,
// a wave's tile is one contiguous 1KB block (fully coalesced global loads).
__device__ __forceinline__ v16h load_B_pk(const _Float16* __restrict__ buf,
                                          int kk, int n0, int N, int lane) {
  const _Float16* p =
      buf + ((size_t)kk * N + n0 + (lane & 15)) * 32 + 16 * (lane >> 4);
  v8h a = *(const v8h*)(p);
  v8h b = *(const v8h*)(p + 8);
  v16h r;
#pragma unroll
  for (int e = 0; e < 8; ++e) { r[e] = a[e]; r[e + 8] = b[e]; }
  return r;
}

// Store f32 D tile (n=lane%16, m=r+8*(lane/16)) as f16, row-major.
__device__ __forceinline__ void store_D_f16(_Float16* buf, v8f acc, int m0,
                                            int n0, int ld, int lane) {
  int n = n0 + (lane & 15);
  int h = lane >> 4;
#pragma unroll
  for (int r = 0; r < 8; ++r) buf[(m0 + r + 8 * h) * ld + n] = (_Float16)acc[r];
}

// Store f32 D tile transposed: buf[n][m] (for vT so P@V B-loads are contiguous)
__device__ __forceinline__ void store_D_f16_T(_Float16* buf, v8f acc, int m0,
                                              int n0, int ld, int lane) {
  int n = n0 + (lane & 15);
  int h = lane >> 4;
#pragma unroll
  for (int r = 0; r < 8; ++r) buf[n * ld + m0 + r + 8 * h] = (_Float16)acc[r];
}

__device__ __forceinline__ float half16_max(float v) {
#pragma unroll
  for (int m = 1; m < 16; m <<= 1) v = fmaxf(v, __shfl_xor(v, m, 32));
  return v;
}
__device__ __forceinline__ float half16_sum(float v) {
#pragma unroll
  for (int m = 1; m < 16; m <<= 1) v += __shfl_xor(v, m, 32);
  return v;
}
__device__ __forceinline__ float wave_sum(float v) {
#pragma unroll
  for (int m = 1; m < 32; m <<= 1) v += __shfl_xor(v, m, 32);
  return v;
}

// LayerNorm over C=256: wave w handles rows w*8..w*8+7; f16 pairs packed as
// 4-byte stores (v_cvt_pk_f16_f32 + ds_store_b32).
__device__ __forceinline__ void layernorm_rows(const float* X, _Float16* Hout,
                                               const float* __restrict__ g,
                                               const float* __restrict__ bb,
                                               int w, int lane) {
#pragma unroll 1
  for (int r = 0; r < 8; ++r) {
    const int t = w * 8 + r;
    const float* xr = X + t * CC;
    float xv[8];
    float s = 0.f;
#pragma unroll
    for (int j = 0; j < 4; ++j) {
      const int c = 64 * j + 2 * lane;
      xv[2 * j] = xr[c];
      xv[2 * j + 1] = xr[c + 1];
      s += xv[2 * j] + xv[2 * j + 1];
    }
    const float mu = wave_sum(s) * (1.0f / CC);
    float v = 0.f;
#pragma unroll
    for (int j = 0; j < 8; ++j) { float d = xv[j] - mu; v += d * d; }
    const float rs = rsqrtf(wave_sum(v) * (1.0f / CC) + LN_EPS);
#pragma unroll
    for (int j = 0; j < 4; ++j) {
      const int c = 64 * j + 2 * lane;
      v2h p;
      p[0] = (_Float16)((xv[2 * j] - mu) * rs * g[c] + bb[c]);
      p[1] = (_Float16)((xv[2 * j + 1] - mu) * rs * g[c + 1] + bb[c + 1]);
      *(v2h*)(Hout + t * CC + c) = p;
    }
  }
}

__global__ __launch_bounds__(256) void transformer_block_kernel(
    const float* __restrict__ x, const _Float16* __restrict__ wpk,
    const float* __restrict__ bp, const float* __restrict__ b1,
    const float* __restrict__ b2, const float* __restrict__ g1,
    const float* __restrict__ be1, const float* __restrict__ g2,
    const float* __restrict__ be2, float* __restrict__ out) {
  __shared__ float    Xs[TT * CC];       // 64 KB: x, then x+sa (residual)
  __shared__ _Float16 Hs[TT * CC];       // 32 KB: LN1 out, later LN2 out
  __shared__ _Float16 Os[TT * NH * HD];  // 64 KB: concat attention output
  __shared__ _Float16 Scr[8 * 6144];     // 96 KB: per-wave q/k/vT (+P); FFN G

  const int b = blockIdx.x;
  const int tid = threadIdx.x;
  const int w = tid >> 5, lane = tid & 31;
  const float* xg = x + (size_t)b * TT * CC;

  // ---- stage x ----
  for (int i = tid; i < TT * CC / 4; i += 256)
    ((float4*)Xs)[i] = ((const float4*)xg)[i];
  __syncthreads();

  // ---- LN1 ----
  layernorm_rows(Xs, Hs, g1, be1, w, lane);
  __syncthreads();

  _Float16* qw = Scr + w * 6144;  // 64x32 f16
  _Float16* kw = qw + 2048;       // 64x32 f16
  _Float16* vw = qw + 4096;       // vT: 32x64 f16
  _Float16* Pw = qw;              // 64x64 f16, overlays q,k after scores

  // ---- attention: each wave owns heads w and w+8 ----
#pragma unroll 1
  for (int it = 0; it < 2; ++it) {
    const int hd = w + it * 8;
    // Direct pointers (no pointer-array select -> stays global_load).
    const _Float16* __restrict__ WqPh = wpk + WQP_OFF + (size_t)hd * 8192;
    const _Float16* __restrict__ WkPh = wpk + WKP_OFF + (size_t)hd * 8192;
    const _Float16* __restrict__ WvPh = wpk + WVP_OFF + (size_t)hd * 8192;

    // q/k/v = Hs(64x256) @ W(256x32): A loaded per k-step (low reg pressure),
    // shared across the two N-tiles; q/k/v sequential.
#pragma unroll 1
    for (int mt = 0; mt < 4; ++mt) {
      auto qkv_mat = [&](const _Float16* __restrict__ Wb, _Float16* dst,
                         bool transT) {
        v8f a0 = {}, a1 = {};
#pragma unroll
        for (int kk = 0; kk < 8; ++kk) {
          v16h A = load_A_lds(Hs, mt * 16, kk * 32, CC, lane);
          a0 = wmma16(A, load_B_pk(Wb, kk, 0, HD, lane), a0);
          a1 = wmma16(A, load_B_pk(Wb, kk, 16, HD, lane), a1);
        }
        if (transT) {
          store_D_f16_T(dst, a0, mt * 16, 0, TT, lane);   // vT[d][s]
          store_D_f16_T(dst, a1, mt * 16, 16, TT, lane);
        } else {
          store_D_f16(dst, a0, mt * 16, 0, HD, lane);
          store_D_f16(dst, a1, mt * 16, 16, HD, lane);
        }
      };
      qkv_mat(WqPh, qw, false);
      qkv_mat(WkPh, kw, false);
      qkv_mat(WvPh, vw, true);
    }

    // scores = q @ k^T (K = 32 = head_size: one WMMA per tile); A hoisted
    v8f st[4][4];
#pragma unroll
    for (int mt = 0; mt < 4; ++mt) {
      v16h A = load_A_lds(qw, mt * 16, 0, HD, lane);
#pragma unroll
      for (int nt = 0; nt < 4; ++nt) {
        v8f a = {};
        if (nt <= mt)
          a = wmma16(A, load_Bt_lds(kw, nt * 16, 0, HD, lane), a);
        st[mt][nt] = a;
      }
    }

    // causal softmax in registers; write P over q/k region
    const int h = lane >> 4, n = lane & 15;
#pragma unroll
    for (int mt = 0; mt < 4; ++mt) {
#pragma unroll
      for (int r = 0; r < 8; ++r) {
        const int t = mt * 16 + 8 * h + r;
        float vals[4];
        float mx = -3.0e38f;
#pragma unroll
        for (int nt = 0; nt < 4; ++nt) {
          int s = nt * 16 + n;
          float vv = (nt <= mt && s <= t) ? st[mt][nt][r] * SCALE_QK : -3.0e38f;
          vals[nt] = vv;
          mx = fmaxf(mx, vv);
        }
        mx = half16_max(mx);
        float sum = 0.f;
#pragma unroll
        for (int nt = 0; nt < 4; ++nt) {
          float e = (vals[nt] > -1.0e38f) ? __expf(vals[nt] - mx) : 0.f;
          vals[nt] = e;
          sum += e;
        }
        sum = half16_sum(sum);
        float inv = 1.0f / sum;
#pragma unroll
        for (int nt = 0; nt < 4; ++nt)
          Pw[t * TT + nt * 16 + n] = (_Float16)(vals[nt] * inv);
      }
    }

    // o = P(64x64) @ v(64x32); vT gives contiguous B reads; skip masked blocks
#pragma unroll 1
    for (int mt = 0; mt < 4; ++mt) {
      const int kkmax = (mt * 16 + 15) / 32;  // 0,0,1,1
      v8f acc[2] = {};
      for (int kk = 0; kk <= kkmax; ++kk) {
        v16h A = load_A_lds(Pw, mt * 16, kk * 32, TT, lane);
#pragma unroll
        for (int nt = 0; nt < 2; ++nt)
          acc[nt] =
              wmma16(A, load_Bt_lds(vw, nt * 16, kk * 32, TT, lane), acc[nt]);
      }
#pragma unroll
      for (int nt = 0; nt < 2; ++nt)
        store_D_f16(Os, acc[nt], mt * 16, hd * HD + nt * 16, NH * HD, lane);
    }
  }
  __syncthreads();

  // ---- projection + residual: wave w owns output columns [w*32, w*32+32) ----
#pragma unroll 1
  for (int mt = 0; mt < 4; ++mt) {
    v8f acc[2] = {};
    for (int kk = 0; kk < 16; ++kk) {  // K = 512
      v16h A = load_A_lds(Os, mt * 16, kk * 32, NH * HD, lane);
#pragma unroll
      for (int i = 0; i < 2; ++i)
        acc[i] = wmma16(
            A, load_B_pk(wpk + WPP_OFF, kk, w * 32 + i * 16, CC, lane), acc[i]);
    }
    const int h = lane >> 4;
#pragma unroll
    for (int i = 0; i < 2; ++i) {
      const int n = w * 32 + i * 16 + (lane & 15);
      const float bpv = bp[n];
#pragma unroll
      for (int r = 0; r < 8; ++r) {
        const int t = mt * 16 + 8 * h + r;
        Xs[t * CC + n] += acc[i][r] + bpv;  // x = x + sa (disjoint cols)
      }
    }
  }
  __syncthreads();

  // ---- LN2 ----
  layernorm_rows(Xs, Hs, g2, be2, w, lane);
  __syncthreads();

  // ---- FFN in 8 blocks of 128 hidden cols; output accum stays in registers --
  _Float16* G = Scr;  // 64x128 f16 relu intermediate (re-uses attention scratch)
  v8f acc2[4][2];
#pragma unroll
  for (int mt = 0; mt < 4; ++mt)
    for (int i = 0; i < 2; ++i) acc2[mt][i] = (v8f){};

#pragma unroll 1
  for (int blk = 0; blk < 8; ++blk) {
    // g = relu(Hs @ W1[:, blk]) ; wave w computes hidden cols w*16..w*16+15
    const int gh0 = blk * 128 + w * 16;
#pragma unroll 1
    for (int mt = 0; mt < 4; ++mt) {
      v8f acc = {};
#pragma unroll
      for (int kk = 0; kk < 8; ++kk)
        acc = wmma16(load_A_lds(Hs, mt * 16, kk * 32, CC, lane),
                     load_B_pk(wpk + W1P_OFF, kk, gh0, FFD, lane), acc);
      const int n = lane & 15, h = lane >> 4;
      const float b1v = b1[gh0 + n];
#pragma unroll
      for (int r = 0; r < 8; ++r)
        G[(mt * 16 + 8 * h + r) * 128 + w * 16 + n] =
            (_Float16)fmaxf(acc[r] + b1v, 0.f);
    }
    __syncthreads();

    // out += g @ W2[blk*128 : blk*128+128, :]
#pragma unroll 1
    for (int mt = 0; mt < 4; ++mt)
      for (int kk = 0; kk < 4; ++kk) {
        v16h A = load_A_lds(G, mt * 16, kk * 32, 128, lane);
#pragma unroll
        for (int i = 0; i < 2; ++i)
          acc2[mt][i] = wmma16(
              A,
              load_B_pk(wpk + W2P_OFF, blk * 4 + kk, w * 32 + i * 16, CC, lane),
              acc2[mt][i]);
      }
    __syncthreads();
  }

  // ---- final residual + bias -> global ----
  float* og = out + (size_t)b * TT * CC;
#pragma unroll 1
  for (int mt = 0; mt < 4; ++mt)
    for (int i = 0; i < 2; ++i) {
      const int n = w * 32 + i * 16 + (lane & 15), h = lane >> 4;
      const float b2v = b2[n];
#pragma unroll
      for (int r = 0; r < 8; ++r) {
        const int t = mt * 16 + 8 * h + r;
        og[t * CC + n] = Xs[t * CC + n] + acc2[mt][i][r] + b2v;
      }
    }
}

extern "C" void kernel_launch(void* const* d_in, const int* in_sizes, int n_in,
                              void* d_out, int out_size, void* d_ws,
                              size_t ws_size, hipStream_t stream) {
  (void)in_sizes; (void)n_in; (void)ws_size; (void)out_size;
  const float* x   = (const float*)d_in[0];
  const float* Wq  = (const float*)d_in[1];
  const float* Wk  = (const float*)d_in[2];
  const float* Wv  = (const float*)d_in[3];
  const float* Wp  = (const float*)d_in[4];
  const float* bp  = (const float*)d_in[5];
  const float* W1  = (const float*)d_in[6];
  const float* b1  = (const float*)d_in[7];
  const float* W2  = (const float*)d_in[8];
  const float* b2  = (const float*)d_in[9];
  const float* g1  = (const float*)d_in[10];
  const float* be1 = (const float*)d_in[11];
  const float* g2  = (const float*)d_in[12];
  const float* be2 = (const float*)d_in[13];
  float* out = (float*)d_out;
  _Float16* wpk = (_Float16*)d_ws;  // needs 2 MB

  // Pre-pack all weights to f16 B-tile layout (runs every call; idempotent).
  pack_weights_kernel<<<dim3(8, NH), 256, 0, stream>>>(
      Wq, wpk + WQP_OFF, CC, HD, (size_t)CC * HD, 8192);
  pack_weights_kernel<<<dim3(8, NH), 256, 0, stream>>>(
      Wk, wpk + WKP_OFF, CC, HD, (size_t)CC * HD, 8192);
  pack_weights_kernel<<<dim3(8, NH), 256, 0, stream>>>(
      Wv, wpk + WVP_OFF, CC, HD, (size_t)CC * HD, 8192);
  pack_weights_kernel<<<dim3(64, 1), 256, 0, stream>>>(
      Wp, wpk + WPP_OFF, NH * HD, CC, 0, 0);
  pack_weights_kernel<<<dim3(128, 1), 256, 0, stream>>>(
      W1, wpk + W1P_OFF, CC, FFD, 0, 0);
  pack_weights_kernel<<<dim3(128, 1), 256, 0, stream>>>(
      W2, wpk + W2P_OFF, FFD, CC, 0, 0);

  transformer_block_kernel<<<2048, 256, 0, stream>>>(
      x, wpk, bp, b1, b2, g1, be1, g2, be2, out);
}
